// LSTMmodel_50294067036179
// MI455X (gfx1250) — matrix operand order repaired and checked by reading, not compiled, not run on hardware
//
#include <hip/hip_runtime.h>

typedef __attribute__((ext_vector_type(16))) _Float16 v16h;
typedef __attribute__((ext_vector_type(8)))  _Float16 v8h;
typedef __attribute__((ext_vector_type(8)))  float    v8f;

#define T_DIM   365
#define B_DIM   1024
#define NX_DIM  16
#define H_DIM   256
#define G_DIM   (4 * H_DIM)              // 1024 gate columns
#define M_TOTAL (T_DIM * B_DIM)          // 373760 rows, divisible by 32

__device__ __forceinline__ float sig_f(float x) {
    return 1.0f / (1.0f + __expf(-x));
}
__device__ __forceinline__ float tanh_f(float x) {
    // tanh(x) = 1 - 2/(e^{2x}+1); saturates correctly at +/-1
    return 1.0f - 2.0f / (__expf(2.0f * x) + 1.0f);
}

// ---------------------------------------------------------------------------
// prep: W_ih -> f16, Win -> f16 padded K 16->32 with zeros, biasg = b_ih+b_hh
// ---------------------------------------------------------------------------
__global__ void prep_kernel(const float* __restrict__ W_ih,
                            const float* __restrict__ Win,
                            const float* __restrict__ b_ih,
                            const float* __restrict__ b_hh,
                            _Float16* __restrict__ wih16,
                            _Float16* __restrict__ winp16,
                            float* __restrict__ biasg) {
    int gid = blockIdx.x * blockDim.x + threadIdx.x;
    if (gid < G_DIM * H_DIM) wih16[gid] = (_Float16)W_ih[gid];
    if (gid < H_DIM * 32) {
        int h = gid >> 5, k = gid & 31;
        winp16[gid] = (k < NX_DIM) ? (_Float16)Win[h * NX_DIM + k] : (_Float16)0.0f;
    }
    if (gid < G_DIM) biasg[gid] = b_ih[gid] + b_hh[gid];
}

// ---------------------------------------------------------------------------
// Fused LSTM-step kernel. Workgroup = 32 rows of M, 6 waves:
//   gi = wid>>1 in {0=i,1=g,2=o} (f-gate skipped: c0==0),
//   nh = wid&1  -> which 128-column half of the gate this wave computes.
// Each wave register-blocks 2 M-strips x 8 N-tiles (acc = 2x8 v8f = 128 VGPR),
// so every B fragment load feeds two WMMAs. K-loop fully unrolled: 128 WMMAs
// in a straight line so the scheduler can bury B/A loads under the XDL work.
// ---------------------------------------------------------------------------
__global__ void __launch_bounds__(192) lstm_fused_kernel(
    const float* __restrict__ x,
    const _Float16* __restrict__ winp16,
    const float* __restrict__ b_in,
    const _Float16* __restrict__ wih16,
    const float* __restrict__ biasg,
    const float* __restrict__ Wout,
    const float* __restrict__ b_out,
    float* __restrict__ y) {
    __shared__ _Float16 x0s[2][16][H_DIM];     // 16 KB, stage-1 result (f16)
    __shared__ float    lds_si[2][16][H_DIM];  // 32 KB, sigmoid(i)
    __shared__ float    lds_tg[2][16][H_DIM];  // 32 KB, tanh(g)
    __shared__ float    ypar[2][32];           // per-half y partials

    const int wid     = threadIdx.x >> 5;      // 0..5
    const int lane    = threadIdx.x & 31;
    const int nlane   = lane & 15;             // N (or M-row) index within tile
    const int hi_half = (lane >= 16) ? 1 : 0;
    const int mloc    = hi_half ? 8 : 0;       // C/D: M = vgpr + (lane<16?0:8)
    const int gi      = wid >> 1;              // 0=i, 1=g, 2=o
    const int nh      = wid & 1;               // column half of the gate
    const int mbase   = blockIdx.x * 32;

    // -------- stage 1: x0 tile (32x256) via WMMA, result into LDS ----------
    // A 16x32 f16 layout: lane<16 holds M=lane, K{0..7,16..23}; lane>=16 holds
    // K{8..15,24..31}. Real K is 0..15 (NX), 16..31 padded zero.
    for (int tt = wid; tt < 32; tt += 6) {
        const int s   = tt >> 4;               // strip
        const int n   = tt & 15;               // 16-column tile of H
        const int row = mbase + s * 16 + nlane;
        const float* xp = x + (size_t)row * NX_DIM + (hi_half ? 8 : 0);
        v16h a;
#pragma unroll
        for (int e = 0; e < 8; ++e) a[e] = (_Float16)xp[e];
#pragma unroll
        for (int e = 8; e < 16; ++e) a[e] = (_Float16)0.0f;

        const int h = n * 16 + nlane;          // output column (fixed per lane)
        v8h blo = *(const v8h*)(winp16 + h * 32 + (hi_half ? 8 : 0));
        v8h bhi = *(const v8h*)(winp16 + h * 32 + 16 + (hi_half ? 8 : 0));
        v16h b;
#pragma unroll
        for (int e = 0; e < 8; ++e) { b[e] = blo[e]; b[8 + e] = bhi[e]; }
        v8f c = {0.f, 0.f, 0.f, 0.f, 0.f, 0.f, 0.f, 0.f};
        c = __builtin_amdgcn_wmma_f32_16x16x32_f16(false, a, false, b,
                                                   (short)0, c, false, false);
        const float bias = b_in[h];
#pragma unroll
        for (int v = 0; v < 8; ++v) {
            float r = c[v] + bias;
            r = r > 0.f ? r : 0.f;
            x0s[s][mloc + v][h] = (_Float16)r;
        }
    }
    __syncthreads();

    // -------- stage 2: gate GEMM, 2 strips x 8 n-tiles per wave ------------
    const int gbase = (gi == 0) ? 0 : ((gi == 1) ? 2 * H_DIM : 3 * H_DIM);
    v8f acc[2][8];
#pragma unroll
    for (int s = 0; s < 2; ++s)
#pragma unroll
        for (int j = 0; j < 8; ++j) {
            v8f z = {0.f, 0.f, 0.f, 0.f, 0.f, 0.f, 0.f, 0.f};
            acc[s][j] = z;
        }

    const int aoff  = hi_half ? 16 : 0;
    const int khoff = hi_half ? 16 : 0;
    // base of this wave's B columns: gate + half + per-lane column
    const _Float16* bbase =
        wih16 + (size_t)(gbase + nh * 128 + nlane) * H_DIM + khoff;

#pragma unroll
    for (int k0 = 0; k0 < H_DIM; k0 += 32) {
        v16h a[2];
#pragma unroll
        for (int s = 0; s < 2; ++s) {
            v8h alo = *(const v8h*)(&x0s[s][nlane][k0 + aoff]);
            v8h ahi = *(const v8h*)(&x0s[s][nlane][k0 + 8 + aoff]);
#pragma unroll
            for (int e = 0; e < 8; ++e) { a[s][e] = alo[e]; a[s][8 + e] = ahi[e]; }
        }
#pragma unroll
        for (int j = 0; j < 8; ++j) {
            v16h b = *(const v16h*)(bbase + (size_t)j * 16 * H_DIM + k0);
            acc[0][j] = __builtin_amdgcn_wmma_f32_16x16x32_f16(
                false, a[0], false, b, (short)0, acc[0][j], false, false);
            acc[1][j] = __builtin_amdgcn_wmma_f32_16x16x32_f16(
                false, a[1], false, b, (short)0, acc[1][j], false, false);
        }
    }

    // -------- stage 3: activations, cross-gate combine, y = h1 . Wout ------
    if (gi == 0) {
#pragma unroll
        for (int s = 0; s < 2; ++s)
#pragma unroll
            for (int j = 0; j < 8; ++j) {
                const int h = (nh * 8 + j) * 16 + nlane;
                const float bb = biasg[h];
#pragma unroll
                for (int v = 0; v < 8; ++v)
                    lds_si[s][mloc + v][h] = sig_f(acc[s][j][v] + bb);
            }
    } else if (gi == 1) {
#pragma unroll
        for (int s = 0; s < 2; ++s)
#pragma unroll
            for (int j = 0; j < 8; ++j) {
                const int h = (nh * 8 + j) * 16 + nlane;
                const float bb = biasg[2 * H_DIM + h];
#pragma unroll
                for (int v = 0; v < 8; ++v)
                    lds_tg[s][mloc + v][h] = tanh_f(acc[s][j][v] + bb);
            }
    }
    __syncthreads();

    if (gi == 2) {
        float ypart[2][8];
#pragma unroll
        for (int s = 0; s < 2; ++s)
#pragma unroll
            for (int v = 0; v < 8; ++v) ypart[s][v] = 0.f;

#pragma unroll
        for (int s = 0; s < 2; ++s)
#pragma unroll
            for (int j = 0; j < 8; ++j) {
                const int h = (nh * 8 + j) * 16 + nlane;
                const float bb = biasg[3 * H_DIM + h];
                const float w = Wout[h];
#pragma unroll
                for (int v = 0; v < 8; ++v) {
                    float c1 = lds_si[s][mloc + v][h] * lds_tg[s][mloc + v][h];
                    float h1 = sig_f(acc[s][j][v] + bb) * tanh_f(c1);
                    ypart[s][v] += h1 * w;
                }
            }
        // reduce across the 16 lanes that share each output row
#pragma unroll
        for (int off = 8; off >= 1; off >>= 1) {
#pragma unroll
            for (int s = 0; s < 2; ++s)
#pragma unroll
                for (int v = 0; v < 8; ++v)
                    ypart[s][v] += __shfl_down(ypart[s][v], off, 16);
        }
        if (nlane == 0) {
#pragma unroll
            for (int s = 0; s < 2; ++s)
#pragma unroll
                for (int v = 0; v < 8; ++v)
                    ypar[nh][s * 16 + mloc + v] = ypart[s][v];
        }
    }
    __syncthreads();

    // sum the two column-halves, add bias, write 32 rows
    if (threadIdx.x < 32) {
        const int row = threadIdx.x;
        y[mbase + row] = ypar[0][row] + ypar[1][row] + b_out[0];
    }
}

// ---------------------------------------------------------------------------
extern "C" void kernel_launch(void* const* d_in, const int* in_sizes, int n_in,
                              void* d_out, int out_size, void* d_ws, size_t ws_size,
                              hipStream_t stream) {
    const float* x     = (const float*)d_in[0];  // [T,B,NX]
    const float* Win   = (const float*)d_in[1];  // [H,NX]
    const float* b_in  = (const float*)d_in[2];  // [H]
    const float* W_ih  = (const float*)d_in[3];  // [4H,H]
    const float* b_ih  = (const float*)d_in[4];  // [4H]
    const float* b_hh  = (const float*)d_in[5];  // [4H]
    const float* Wout  = (const float*)d_in[6];  // [NY,H], NY=1
    const float* b_out = (const float*)d_in[7];  // [NY]
    float* y = (float*)d_out;                    // [T,B,1]

    unsigned char* ws = (unsigned char*)d_ws;
    const size_t WIH_BYTES  = (size_t)G_DIM * H_DIM * sizeof(_Float16); // 512 KB
    const size_t WINP_BYTES = (size_t)H_DIM * 32 * sizeof(_Float16);    // 16 KB
    _Float16* wih16  = (_Float16*)ws;
    _Float16* winp16 = (_Float16*)(ws + WIH_BYTES);
    float*    biasg  = (float*)(ws + WIH_BYTES + WINP_BYTES);

    prep_kernel<<<(G_DIM * H_DIM + 255) / 256, 256, 0, stream>>>(
        W_ih, Win, b_ih, b_hh, wih16, winp16, biasg);

    lstm_fused_kernel<<<M_TOTAL / 32, 192, 0, stream>>>(
        x, winp16, b_in, wih16, biasg, Wout, b_out, y);
}
